// TransformerBlock_36515811950995
// MI455X (gfx1250) — matrix operand order repaired and checked by reading, not compiled, not run on hardware
//
#include <hip/hip_runtime.h>

typedef __bf16 bf16_t;
typedef __attribute__((ext_vector_type(16))) __bf16 v16bf;
typedef __attribute__((ext_vector_type(8)))  __bf16 v8bf;
typedef __attribute__((ext_vector_type(8)))  float  v8f;

#define BQ   2
#define SQ   2048
#define DQ   1024
#define HQ   16
#define HDQ  64
#define FFQ  4096
#define MSQ  (BQ*SQ)   // 4096 rows

// ---------------------------------------------------------------- helpers
__device__ inline v8f zero8() {
  v8f z;
#pragma unroll
  for (int i = 0; i < 8; ++i) z[i] = 0.0f;
  return z;
}

__device__ inline v16bf combine16(v8bf lo, v8bf hi) {
  v16bf r;
#pragma unroll
  for (int i = 0; i < 8; ++i) { r[i] = lo[i]; r[i + 8] = hi[i]; }
  return r;
}

__device__ inline v8f wmma_bf16f32(v16bf a, v16bf b, v8f c) {
  return __builtin_amdgcn_wmma_f32_16x16x32_bf16(false, a, false, b, (short)0, c,
                                                 false, false);
}

// A-matrix fragment (16x32 bf16): K-contiguous source, row stride `ld` elems.
// half0 lanes: k {0-7,16-23}; half1 lanes: k {8-15,24-31}
__device__ inline v16bf load_a_frag(const bf16_t* __restrict__ p, int lane, int ld) {
  const bf16_t* base = p + (size_t)(lane & 15) * ld + ((lane < 16) ? 0 : 8);
  return combine16(*(const v8bf*)(base), *(const v8bf*)(base + 16));
}

// B-matrix fragment (32x16 bf16) from K-contiguous (transposed) source,
// col stride `ld` elems. half0 lanes: k {0-7,16-23}; half1 lanes: k {8-15,24-31}
__device__ inline v16bf load_b_frag(const bf16_t* __restrict__ p, int lane, int ld) {
  const bf16_t* base = p + (size_t)(lane & 15) * ld + ((lane < 16) ? 0 : 8);
  return combine16(*(const v8bf*)(base), *(const v8bf*)(base + 16));
}

// ---------------------------------------------------------------- fp32 -> bf16
__global__ __launch_bounds__(256)
void cvt_f32_bf16(const float* __restrict__ in, bf16_t* __restrict__ out, int n) {
  int i = blockIdx.x * 256 + threadIdx.x;
  int stride = gridDim.x * 256;
  for (; i < n; i += stride) out[i] = (bf16_t)in[i];
}

// ---------------------------------------------------------------- fp32 [K,N] -> bf16 [N,K]
__global__ __launch_bounds__(256)
void cvt_transpose(const float* __restrict__ W, bf16_t* __restrict__ Wt, int K, int N) {
  __shared__ float tile[32][33];
  int n0 = blockIdx.x * 32, k0 = blockIdx.y * 32;
  int c = threadIdx.x & 31, rg = threadIdx.x >> 5;
#pragma unroll
  for (int i = 0; i < 4; ++i) {
    int k = rg + i * 8;
    tile[k][c] = W[(size_t)(k0 + k) * N + n0 + c];
  }
  __syncthreads();
#pragma unroll
  for (int i = 0; i < 4; ++i) {
    int n = rg + i * 8;
    Wt[(size_t)(n0 + n) * K + k0 + c] = (bf16_t)tile[c][n];
  }
}

// ---------------------------------------------------------------- mask -> 32-key bitmask words
__global__ __launch_bounds__(256)
void pack_mask(const int* __restrict__ mask, unsigned* __restrict__ maskp) {
  int row  = blockIdx.x * 8 + (threadIdx.x >> 5);
  int lane = threadIdx.x & 31;
  const int* mrow = mask + (size_t)row * SQ;
  for (int b = 0; b < SQ / 32; ++b) {
    unsigned w = (unsigned)__ballot(mrow[b * 32 + lane] != 0);
    if (lane == 0) maskp[(size_t)row * (SQ / 32) + b] = w;
  }
}

// ---------------------------------------------------------------- GEMM, LDS-free
// C[M,N] = A[M,K] @ W[K,N], with W supplied pre-transposed: Wt[N,K] (bf16).
// MODE 0: (acc+bias)*scale -> bf16        (Q/K; Q scale=1/8)
// MODE 1: acc+bias+resid   -> f32         (O-proj + residual x)
// MODE 2: gelu(acc+bias)   -> bf16        (FFN1)
// MODE 3: acc+bias+resid   -> f32         (FFN2 + residual h1)
// MODE 4: (acc+bias)       -> bf16, stored head-transposed (B*H, 64, S)  (V)
template <int MODE>
__global__ __launch_bounds__(256)
void gemm_bf16(const bf16_t* __restrict__ A, const bf16_t* __restrict__ Wt,
               const float* __restrict__ bias, const float* __restrict__ resid,
               float scale, int M, int N, int K,
               bf16_t* __restrict__ outb, float* __restrict__ outf) {
  const int tid   = threadIdx.x;
  const int lane  = tid & 31;
  const int wave  = tid >> 5;
  const int wr    = wave >> 1;          // 0..3 : 32-row band
  const int wc    = wave & 1;           // 0..1 : 64-col band
  const int m0    = blockIdx.y * 128 + wr * 32;
  const int n0    = blockIdx.x * 128 + wc * 64;
  const int l15   = lane & 15;
  const int half8 = (lane < 16) ? 0 : 8;

  v8f acc[2][4];
#pragma unroll
  for (int i = 0; i < 2; ++i)
#pragma unroll
    for (int j = 0; j < 4; ++j) acc[i][j] = zero8();

  const bf16_t* Ab = A  + (size_t)m0 * K;
  const bf16_t* Wb = Wt + (size_t)n0 * K;

  for (int k0 = 0; k0 < K; k0 += 32) {
    if (k0 + 32 < K) {  // global_prefetch for the streamed K-direction
      __builtin_prefetch(Ab + (size_t)l15 * K + k0 + 32, 0, 1);
      __builtin_prefetch(Wb + (size_t)l15 * K + k0 + 32, 0, 1);
    }
    v16bf af[2];
#pragma unroll
    for (int rt = 0; rt < 2; ++rt)
      af[rt] = load_a_frag(Ab + (size_t)rt * 16 * K + k0, lane, K);
    v16bf bfm[4];
#pragma unroll
    for (int ct = 0; ct < 4; ++ct)
      bfm[ct] = load_b_frag(Wb + (size_t)ct * 16 * K + k0, lane, K);
#pragma unroll
    for (int rt = 0; rt < 2; ++rt)
#pragma unroll
      for (int ct = 0; ct < 4; ++ct)
        acc[rt][ct] = wmma_bf16f32(af[rt], bfm[ct], acc[rt][ct]);
  }

  // epilogue (C layout: row = vgpr + 8*half, col = lane%16)
#pragma unroll
  for (int rt = 0; rt < 2; ++rt) {
#pragma unroll
    for (int ct = 0; ct < 4; ++ct) {
      int col = n0 + ct * 16 + l15;
      float bv = bias[col];
      int rowbase = m0 + rt * 16 + half8;
#pragma unroll
      for (int r = 0; r < 8; ++r) {
        int row = rowbase + r;
        float v = acc[rt][ct][r] + bv;
        if (MODE == 0) {
          outb[(size_t)row * N + col] = (bf16_t)(v * scale);
        } else if (MODE == 2) {
          float g = 0.5f * v * (1.0f + erff(v * 0.70710678118654752f));
          outb[(size_t)row * N + col] = (bf16_t)g;
        } else if (MODE == 4) {
          // store V as (B*H, 64, S): faithful-reshape head layout, transposed
          int b = row >> 11, s = row & (SQ - 1);
          int bh = b * HQ + (s >> 7);
          int t  = ((s & 127) << 4) + (col >> 6);
          int e  = col & 63;
          outb[(size_t)bh * (HDQ * SQ) + (size_t)e * SQ + t] = (bf16_t)v;
        } else {
          size_t idx = (size_t)row * N + col;
          outf[idx] = v + resid[idx];
        }
      }
    }
  }
}

// ---------------------------------------------------------------- flash attention
// grid: (S/64 query tiles, B*H). 128 threads = 4 waves, 16 query rows/wave.
// Q,K: (B*H, S, 64) bf16. Vt: (B*H, 64, S) bf16. maskp: packed bitmask words.
// No barriers; only wave-private LDS for the P re-layout.
__global__ __launch_bounds__(128)
void flash_attn(const bf16_t* __restrict__ Q, const bf16_t* __restrict__ Kmat,
                const bf16_t* __restrict__ Vt, const unsigned* __restrict__ maskp,
                bf16_t* __restrict__ out) {
  __shared__ __align__(16) bf16_t Pl[4][16][80];  // per-wave P (16x64, pad 80)

  const int lane  = threadIdx.x & 31;
  const int wave  = threadIdx.x >> 5;
  const int l15   = lane & 15;
  const int half8 = (lane < 16) ? 0 : 8;
  const int bh    = blockIdx.y;
  const int q0    = blockIdx.x * 64 + wave * 16;

  const bf16_t* Qh = Q    + (size_t)bh * SQ * HDQ;
  const bf16_t* Kh = Kmat + (size_t)bh * SQ * HDQ;
  const bf16_t* Vh = Vt   + (size_t)bh * HDQ * SQ;

  // Q fragments: 16 rows x 64 ch = 2 A-fragments (Q pre-scaled by 1/8)
  v16bf qf[2];
#pragma unroll
  for (int f = 0; f < 2; ++f)
    qf[f] = load_a_frag(Qh + (size_t)q0 * HDQ + f * 32, lane, HDQ);

  v8f av[4];
#pragma unroll
  for (int g = 0; g < 4; ++g) av[g] = zero8();
  float rowmax[8], rowsum[8];
#pragma unroll
  for (int r = 0; r < 8; ++r) { rowmax[r] = -1e30f; rowsum[r] = 0.0f; }

  for (int kb = 0; kb < SQ / 64; ++kb) {
    // --- scores: 4 tiles (16q x 16k), K rows read directly as B fragments
    v8f st[4];
#pragma unroll
    for (int T = 0; T < 4; ++T) {
      v8f z = zero8();
#pragma unroll
      for (int f = 0; f < 2; ++f)
        z = wmma_bf16f32(qf[f],
                         load_b_frag(Kh + (size_t)(kb * 64 + T * 16) * HDQ + f * 32,
                                     lane, HDQ),
                         z);
      st[T] = z;
    }

    // --- masked online softmax; each row lives in one 16-lane half
#pragma unroll
    for (int r = 0; r < 8; ++r) {
      int qr = q0 + r + half8;
      unsigned w0 = maskp[(size_t)qr * (SQ / 32) + 2 * kb];
      unsigned w1 = maskp[(size_t)qr * (SQ / 32) + 2 * kb + 1];
      bool mv[4] = { ((w0 >> l15) & 1u) != 0u, ((w0 >> (16 + l15)) & 1u) != 0u,
                     ((w1 >> l15) & 1u) != 0u, ((w1 >> (16 + l15)) & 1u) != 0u };
      float bm = -1e30f;
#pragma unroll
      for (int T = 0; T < 4; ++T) bm = fmaxf(bm, mv[T] ? st[T][r] : -1e30f);
#pragma unroll
      for (int d = 1; d < 16; d <<= 1) bm = fmaxf(bm, __shfl_xor(bm, d, 32));
      float nm = fmaxf(rowmax[r], bm);
      float scalef, p[4];
      if (nm < -1e29f) {            // nothing unmasked yet
        scalef = 1.0f;
#pragma unroll
        for (int T = 0; T < 4; ++T) p[T] = 0.0f;
      } else {
        scalef = __expf(rowmax[r] - nm);
#pragma unroll
        for (int T = 0; T < 4; ++T) p[T] = mv[T] ? __expf(st[T][r] - nm) : 0.0f;
      }
      rowmax[r] = nm;
      float ps = (p[0] + p[1]) + (p[2] + p[3]);
#pragma unroll
      for (int d = 1; d < 16; d <<= 1) ps += __shfl_xor(ps, d, 32);
      rowsum[r] = rowsum[r] * scalef + ps;
#pragma unroll
      for (int g = 0; g < 4; ++g) av[g][r] *= scalef;
#pragma unroll
      for (int T = 0; T < 4; ++T)
        Pl[wave][r + half8][T * 16 + l15] = (bf16_t)p[T];
    }

    // --- AV += P(16x64) @ V(64x64): P via wave-private LDS re-layout,
    //     V fragments contiguous from head-transposed Vt
    v16bf pf[2];
#pragma unroll
    for (int f = 0; f < 2; ++f)
      pf[f] = load_a_frag(&Pl[wave][0][f * 32], lane, 80);
#pragma unroll
    for (int g = 0; g < 4; ++g) {
#pragma unroll
      for (int f = 0; f < 2; ++f)
        av[g] = wmma_bf16f32(pf[f],
                             load_b_frag(Vh + (size_t)(g * 16) * SQ + kb * 64 + f * 32,
                                         lane, SQ),
                             av[g]);
    }
  }

  // --- finalize: divide by row sums, store bf16 (== (B,S,H*HD) reshape layout)
#pragma unroll
  for (int r = 0; r < 8; ++r) {
    int qr = q0 + r + half8;
    float inv = 1.0f / fmaxf(rowsum[r], 1e-30f);
#pragma unroll
    for (int g = 0; g < 4; ++g)
      out[((size_t)bh * SQ + qr) * HDQ + g * 16 + l15] = (bf16_t)(av[g][r] * inv);
  }
}

// ---------------------------------------------------------------- LayerNorm (row per block)
__global__ __launch_bounds__(256)
void layernorm_kernel(const float* __restrict__ in, const float* __restrict__ gamma,
                      const float* __restrict__ beta, float* __restrict__ outf,
                      bf16_t* __restrict__ outb, int D) {
  int row = blockIdx.x;
  const float* x = in + (size_t)row * D;
  float vals[4];
  float s = 0.0f, s2 = 0.0f;
#pragma unroll
  for (int i = 0; i < 4; ++i) {
    float v = x[threadIdx.x + i * 256];
    vals[i] = v; s += v; s2 += v * v;
  }
#pragma unroll
  for (int d = 1; d < 32; d <<= 1) { s += __shfl_xor(s, d, 32); s2 += __shfl_xor(s2, d, 32); }
  __shared__ float rs[8], rs2[8];
  int wave = threadIdx.x >> 5, lane = threadIdx.x & 31;
  if (lane == 0) { rs[wave] = s; rs2[wave] = s2; }
  __syncthreads();
  if (wave == 0) {
    float a  = (lane < 8) ? rs[lane] : 0.0f;
    float b2 = (lane < 8) ? rs2[lane] : 0.0f;
#pragma unroll
    for (int d = 1; d < 8; d <<= 1) { a += __shfl_xor(a, d, 32); b2 += __shfl_xor(b2, d, 32); }
    if (lane == 0) { rs[0] = a; rs2[0] = b2; }
  }
  __syncthreads();
  float mean = rs[0] / D;
  float var  = rs2[0] / D - mean * mean;
  float inv  = rsqrtf(var + 1e-5f);
#pragma unroll
  for (int i = 0; i < 4; ++i) {
    int c = threadIdx.x + i * 256;
    float v = (vals[i] - mean) * inv * gamma[c] + beta[c];
    outf[(size_t)row * D + c] = v;
    if (outb) outb[(size_t)row * D + c] = (bf16_t)v;
  }
}

// ---------------------------------------------------------------- launch
extern "C" void kernel_launch(void* const* d_in, const int* in_sizes, int n_in,
                              void* d_out, int out_size, void* d_ws, size_t ws_size,
                              hipStream_t stream) {
  (void)in_sizes; (void)n_in; (void)out_size; (void)ws_size;
  const float* x    = (const float*)d_in[0];
  const int*   mask = (const int*)d_in[1];
  const float* Wq = (const float*)d_in[2];  const float* bq  = (const float*)d_in[3];
  const float* Wk = (const float*)d_in[4];  const float* bk  = (const float*)d_in[5];
  const float* Wv = (const float*)d_in[6];  const float* bv  = (const float*)d_in[7];
  const float* Wo = (const float*)d_in[8];  const float* bo  = (const float*)d_in[9];
  const float* W1 = (const float*)d_in[10]; const float* b1  = (const float*)d_in[11];
  const float* W2 = (const float*)d_in[12]; const float* b2  = (const float*)d_in[13];
  const float* g1 = (const float*)d_in[14]; const float* be1 = (const float*)d_in[15];
  const float* g2 = (const float*)d_in[16]; const float* be2 = (const float*)d_in[17];

  char* ws = (char*)d_ws;
  size_t off = 0;
  auto alloc = [&](size_t bytes) {
    void* p = ws + off;
    off = (off + bytes + 255) & ~(size_t)255;
    return p;
  };

  const int MS = MSQ;  // 4096
  bf16_t*   xb    = (bf16_t*)alloc((size_t)MS * DQ * 2);
  bf16_t*   wqt   = (bf16_t*)alloc((size_t)DQ * DQ * 2);    // [N=D][K=D]
  bf16_t*   wkt   = (bf16_t*)alloc((size_t)DQ * DQ * 2);
  bf16_t*   wvt   = (bf16_t*)alloc((size_t)DQ * DQ * 2);
  bf16_t*   wot   = (bf16_t*)alloc((size_t)DQ * DQ * 2);
  bf16_t*   w1t   = (bf16_t*)alloc((size_t)FFQ * DQ * 2);   // [N=FF][K=D]
  bf16_t*   w2t   = (bf16_t*)alloc((size_t)DQ * FFQ * 2);   // [N=D][K=FF]
  unsigned* maskp = (unsigned*)alloc((size_t)SQ * (SQ / 32) * 4);
  bf16_t*   Qb    = (bf16_t*)alloc((size_t)MS * DQ * 2);    // (B*H, S, 64)
  bf16_t*   Kb    = (bf16_t*)alloc((size_t)MS * DQ * 2);    // (B*H, S, 64)
  bf16_t*   Vtb   = (bf16_t*)alloc((size_t)MS * DQ * 2);    // (B*H, 64, S)
  bf16_t*   attb  = (bf16_t*)alloc((size_t)MS * DQ * 2);
  float*    z1    = (float*)alloc((size_t)MS * DQ * 4);
  float*    h1f   = (float*)alloc((size_t)MS * DQ * 4);
  bf16_t*   h1b   = (bf16_t*)alloc((size_t)MS * DQ * 2);
  bf16_t*   ffnb  = (bf16_t*)alloc((size_t)MS * FFQ * 2);
  float*    z2    = (float*)alloc((size_t)MS * DQ * 4);

  // 1) activations -> bf16; weights -> transposed bf16; mask -> bitmask
  cvt_f32_bf16<<<1024, 256, 0, stream>>>(x, xb, MS * DQ);
  cvt_transpose<<<dim3(DQ / 32, DQ / 32),  256, 0, stream>>>(Wq, wqt, DQ, DQ);
  cvt_transpose<<<dim3(DQ / 32, DQ / 32),  256, 0, stream>>>(Wk, wkt, DQ, DQ);
  cvt_transpose<<<dim3(DQ / 32, DQ / 32),  256, 0, stream>>>(Wv, wvt, DQ, DQ);
  cvt_transpose<<<dim3(DQ / 32, DQ / 32),  256, 0, stream>>>(Wo, wot, DQ, DQ);
  cvt_transpose<<<dim3(FFQ / 32, DQ / 32), 256, 0, stream>>>(W1, w1t, DQ, FFQ);
  cvt_transpose<<<dim3(DQ / 32, FFQ / 32), 256, 0, stream>>>(W2, w2t, FFQ, DQ);
  pack_mask<<<SQ / 8, 256, 0, stream>>>(mask, maskp);

  // 2) QKV projections (1/sqrt(64)=0.125 folded into Q; V stored head-transposed)
  dim3 gD(DQ / 128, MS / 128);
  gemm_bf16<0><<<gD, 256, 0, stream>>>(xb, wqt, bq, nullptr, 0.125f, MS, DQ, DQ, Qb,  nullptr);
  gemm_bf16<0><<<gD, 256, 0, stream>>>(xb, wkt, bk, nullptr, 1.0f,   MS, DQ, DQ, Kb,  nullptr);
  gemm_bf16<4><<<gD, 256, 0, stream>>>(xb, wvt, bv, nullptr, 1.0f,   MS, DQ, DQ, Vtb, nullptr);

  // 3) flash attention over (B*H, S, 64) heads
  flash_attn<<<dim3(SQ / 64, BQ * HQ), 128, 0, stream>>>(Qb, Kb, Vtb, maskp, attb);

  // 4) O-projection + residual, LN1
  gemm_bf16<1><<<gD, 256, 0, stream>>>(attb, wot, bo, x, 1.0f, MS, DQ, DQ, nullptr, z1);
  layernorm_kernel<<<MS, 256, 0, stream>>>(z1, g1, be1, h1f, h1b, DQ);

  // 5) FFN: GELU(h1 @ W1 + b1) @ W2 + b2 + h1, LN2 -> d_out
  gemm_bf16<2><<<dim3(FFQ / 128, MS / 128), 256, 0, stream>>>(h1b, w1t, b1, nullptr, 1.0f,
                                                              MS, FFQ, DQ, ffnb, nullptr);
  gemm_bf16<3><<<gD, 256, 0, stream>>>(ffnb, w2t, b2, h1f, 1.0f, MS, DQ, FFQ, nullptr, z2);
  layernorm_kernel<<<MS, 256, 0, stream>>>(z2, g2, be2, (float*)d_out, nullptr, DQ);
}
